// GVPEnsemble_70076686402286
// MI455X (gfx1250) — matrix operand order repaired and checked by reading, not compiled, not run on hardware
//
#include <hip/hip_runtime.h>
#include <hip/hip_bf16.h>

#define CEILDIV(a,b) (((a)+(b)-1)/(b))
#define EPSV 1e-8f
#define NNODES 6000
#define NEDGES 96000

typedef __attribute__((ext_vector_type(16))) _Float16 v16h;
typedef __attribute__((ext_vector_type(8)))  float    v8f;

// ---------------------------------------------------------------------------
// generic kernels
// ---------------------------------------------------------------------------

__global__ void fill_f(float* __restrict__ p, float v, int n) {
  int i = blockIdx.x * blockDim.x + threadIdx.x;
  if (i < n) p[i] = v;
}

// pack fp32 weight W[K][N] (row-major, as in reference) into f16 [N][Kp], zero pad
__global__ void pack_w(const float* __restrict__ w, _Float16* __restrict__ out,
                       int K, int N, int Kp) {
  int i = blockIdx.x * blockDim.x + threadIdx.x;
  if (i >= N * Kp) return;
  int n = i / Kp, k = i - n * Kp;
  out[i] = (k < K) ? (_Float16)w[(size_t)k * N + n] : (_Float16)0.f;
}

// A-fragment loaders: 16-bit A 16x32 layout (contiguous k order, matching B):
// lanes 0-15 hold row M=l15, K in {kb..kb+15}; lanes 16-31: kb += 16.
__device__ __forceinline__ v16h load_afrag_full(const float* __restrict__ arp, int kb) {
  v16h af;
#pragma unroll
  for (int j = 0; j < 16; ++j) {
    int k = kb + j;
    af[j] = (_Float16)arp[k];          // unconditional -> coalescible loads
  }
  return af;
}
__device__ __forceinline__ v16h load_afrag_tail(const float* __restrict__ arp, int kb, int K) {
  v16h af;
#pragma unroll
  for (int j = 0; j < 16; ++j) {
    int k = kb + j;
    float x = arp[k];                  // unconditional load (memory-safe in ws)
    af[j] = (_Float16)((k < K) ? x : 0.f);  // value select only (v_cndmask)
  }
  return af;
}

// C[M,N] = act(A[M,K](f32) @ W[K,N] + bias), W prepacked f16 [N][Kp]
// act: 0=none 1=relu 2=sigmoid 3=elu.  NT = output tiles per wave along N.
template <int NT>
__global__ void __launch_bounds__(128)
wmma_gemm_t(const float* __restrict__ A, int lda,
            const _Float16* __restrict__ Wp, int Kp,
            const float* __restrict__ bias,
            float* __restrict__ C, int ldc,
            int M, int K, int N, int act) {
  int wave = (blockIdx.x * 128 + threadIdx.x) >> 5;
  int ngrp = N / (16 * NT);
  int tiles = ((M + 15) >> 4) * ngrp;
  if (wave >= tiles) return;
  int mt = wave / ngrp, ng = wave - mt * ngrp;
  int lane = threadIdx.x & 31;
  int hid = lane >> 4;          // half-wave id
  int l15 = lane & 15;
  int arow = mt * 16 + l15;
  bool arv = arow < M;
  const float* arp = A + (size_t)(arv ? arow : 0) * lda;

  const _Float16* bb[NT];
  v8f acc[NT];
#pragma unroll
  for (int t = 0; t < NT; ++t) {
    bb[t] = Wp + (size_t)((ng * NT + t) * 16 + l15) * Kp + hid * 16;
#pragma unroll
    for (int r = 0; r < 8; ++r) acc[t][r] = 0.f;
  }

  int kfull = K & ~31;
  for (int k0 = 0; k0 < kfull; k0 += 32) {
    v16h af = load_afrag_full(arp, k0 + hid * 16);
#pragma unroll
    for (int t = 0; t < NT; ++t) {
      v16h bf = *(const v16h*)(bb[t] + k0);
      acc[t] = __builtin_amdgcn_wmma_f32_16x16x32_f16(false, af, false, bf,
                                                      (short)0, acc[t], false, false);
    }
  }
  if (kfull < K) {
    v16h af = load_afrag_tail(arp, kfull + hid * 16, K);
#pragma unroll
    for (int t = 0; t < NT; ++t) {
      v16h bf = *(const v16h*)(bb[t] + kfull);
      acc[t] = __builtin_amdgcn_wmma_f32_16x16x32_f16(false, af, false, bf,
                                                      (short)0, acc[t], false, false);
    }
  }

#pragma unroll
  for (int t = 0; t < NT; ++t) {
    int col = (ng * NT + t) * 16 + l15;
    float bv = bias ? bias[col] : 0.f;
#pragma unroll
    for (int r = 0; r < 8; ++r) {
      int row = mt * 16 + r + hid * 8;   // C layout: VGPR r -> M=r (+8 for hi half)
      if (row < M) {
        float v = acc[t][r] + bv;
        if (act == 1) v = fmaxf(v, 0.f);
        else if (act == 2) v = 1.f / (1.f + __expf(-v));
        else if (act == 3) v = v > 0.f ? v : (__expf(v) - 1.f);
        C[(size_t)row * ldc + col] = v;
      }
    }
  }
}

__global__ void relu_cols(float* __restrict__ X, int ld, int cols, int M) {
  int i = blockIdx.x * blockDim.x + threadIdx.x;
  if (i >= M * cols) return;
  int m = i / cols, j = i - m * cols;
  float* p = X + (size_t)m * ld + j;
  *p = fmaxf(*p, 0.f);
}

__global__ void copy_cols(const float* __restrict__ S, int lds,
                          float* __restrict__ D, int ldd, int cols, int M) {
  int i = blockIdx.x * blockDim.x + threadIdx.x;
  if (i >= M * cols) return;
  int m = i / cols, j = i - m * cols;
  D[(size_t)m * ldd + j] = S[(size_t)m * lds + j];
}

// ---------------------------------------------------------------------------
// GVP vector-path kernels
// ---------------------------------------------------------------------------

// Vh[m][h][3] = Vin[m][vi][3] @ wh[vi][h];  X[m][off+h] = ||Vh||
__global__ void vec_in(const float* __restrict__ Vin, int vi,
                       const float* __restrict__ wh, int h,
                       float* __restrict__ Vh, float* __restrict__ X,
                       int ldx, int off, int M) {
  int i = blockIdx.x * blockDim.x + threadIdx.x;
  if (i >= M * h) return;
  int m = i / h, hh = i - m * h;
  const float* vp = Vin + (size_t)m * vi * 3;
  float a0 = 0, a1 = 0, a2 = 0;
  for (int v = 0; v < vi; ++v) {
    float w = wh[v * h + hh];
    a0 += vp[v * 3 + 0] * w; a1 += vp[v * 3 + 1] * w; a2 += vp[v * 3 + 2] * w;
  }
  size_t b = (size_t)m * h * 3 + hh * 3;
  Vh[b] = a0; Vh[b + 1] = a1; Vh[b + 2] = a2;
  X[(size_t)m * ldx + off + hh] = sqrtf(fmaxf(a0 * a0 + a1 * a1 + a2 * a2, EPSV));
}

// Vo[m][vo][3] = gate[m][vo] * (Vh[m][h][3] @ wv[h][vo])
__global__ void vec_out(const float* __restrict__ Vh, int h,
                        const float* __restrict__ gate,
                        const float* __restrict__ wv, int vo,
                        float* __restrict__ Vo, int M) {
  int i = blockIdx.x * blockDim.x + threadIdx.x;
  if (i >= M * vo) return;
  int m = i / vo, v = i - m * vo;
  const float* vp = Vh + (size_t)m * h * 3;
  float a0 = 0, a1 = 0, a2 = 0;
  for (int hh = 0; hh < h; ++hh) {
    float w = wv[hh * vo + v];
    a0 += vp[hh * 3 + 0] * w; a1 += vp[hh * 3 + 1] * w; a2 += vp[hh * 3 + 2] * w;
  }
  float g = gate[(size_t)m * vo + v];
  size_t b = (size_t)m * vo * 3 + v * 3;
  Vo[b] = a0 * g; Vo[b + 1] = a1 * g; Vo[b + 2] = a2 * g;
}

// ---------------------------------------------------------------------------
// embeddings
// ---------------------------------------------------------------------------

// node: s26=[node_s, W_s[seq]] -> LN -> X[:,0:26]; V rms-norm; Vh=V@wh(3x16); X[:,26:42]=||Vh||
__global__ void node_embed_pre(const float* __restrict__ ns_, const float* __restrict__ nv_,
                               const int* __restrict__ seq, const float* __restrict__ Ws,
                               const float* __restrict__ lnb, const float* __restrict__ lng,
                               const float* __restrict__ wh,
                               float* __restrict__ X, float* __restrict__ Vh, int M) {
  int n = blockIdx.x * blockDim.x + threadIdx.x;
  if (n >= M) return;
  float s[26];
  for (int j = 0; j < 6; ++j) s[j] = ns_[n * 6 + j];
  const float* wr = Ws + seq[n] * 20;
  for (int j = 0; j < 20; ++j) s[6 + j] = wr[j];
  float mu = 0; for (int j = 0; j < 26; ++j) mu += s[j]; mu *= (1.f / 26.f);
  float var = 0; for (int j = 0; j < 26; ++j) { float d = s[j] - mu; var += d * d; }
  float rstd = rsqrtf(var * (1.f / 26.f) + 1e-5f);
  for (int j = 0; j < 26; ++j) X[(size_t)n * 42 + j] = (s[j] - mu) * rstd * lng[j] + lnb[j];
  float V[3][3]; float ssum = 0;
  for (int v = 0; v < 3; ++v) {
    float ss = 0;
    for (int c = 0; c < 3; ++c) { float x = nv_[n * 9 + v * 3 + c]; V[v][c] = x; ss += x * x; }
    ssum += fmaxf(ss, EPSV);
  }
  float inv = rsqrtf(ssum * (1.f / 3.f));
  for (int h = 0; h < 16; ++h) {
    float a0 = 0, a1 = 0, a2 = 0;
    for (int v = 0; v < 3; ++v) {
      float w = wh[v * 16 + h];
      a0 += V[v][0] * inv * w; a1 += V[v][1] * inv * w; a2 += V[v][2] * inv * w;
    }
    size_t b = (size_t)n * 48 + h * 3;
    Vh[b] = a0; Vh[b + 1] = a1; Vh[b + 2] = a2;
    X[(size_t)n * 42 + 26 + h] = sqrtf(fmaxf(a0 * a0 + a1 * a1 + a2 * a2, EPSV));
  }
}

__global__ void edge_embed_pre(const float* __restrict__ es_, const float* __restrict__ ev_,
                               const float* __restrict__ lnb, const float* __restrict__ lng,
                               const float* __restrict__ wh,
                               float* __restrict__ X, float* __restrict__ Vh1, int E) {
  int e = blockIdx.x * blockDim.x + threadIdx.x;
  if (e >= E) return;
  float s[32];
  for (int j = 0; j < 32; ++j) s[j] = es_[(size_t)e * 32 + j];
  float mu = 0; for (int j = 0; j < 32; ++j) mu += s[j]; mu *= (1.f / 32.f);
  float var = 0; for (int j = 0; j < 32; ++j) { float d = s[j] - mu; var += d * d; }
  float rstd = rsqrtf(var * (1.f / 32.f) + 1e-5f);
  for (int j = 0; j < 32; ++j) X[(size_t)e * 33 + j] = (s[j] - mu) * rstd * lng[j] + lnb[j];
  float v0 = ev_[e * 3], v1 = ev_[e * 3 + 1], v2 = ev_[e * 3 + 2];
  float ss = fmaxf(v0 * v0 + v1 * v1 + v2 * v2, EPSV);
  float inv = rsqrtf(ss);
  float w = wh[0];
  float a0 = v0 * inv * w, a1 = v1 * inv * w, a2 = v2 * inv * w;
  Vh1[e * 3] = a0; Vh1[e * 3 + 1] = a1; Vh1[e * 3 + 2] = a2;
  X[(size_t)e * 33 + 32] = sqrtf(fmaxf(a0 * a0 + a1 * a1 + a2 * a2, EPSV));
}

__global__ void edge_embed_post(const float* __restrict__ es_out,
                                const float* __restrict__ wsvw, const float* __restrict__ wsvb,
                                const float* __restrict__ Vh1, const float* __restrict__ wv,
                                float* __restrict__ eV, int E) {
  int e = blockIdx.x * blockDim.x + threadIdx.x;
  if (e >= E) return;
  float a = wsvb[0];
  for (int k = 0; k < 32; ++k) a += es_out[(size_t)e * 32 + k] * wsvw[k];
  float g = 1.f / (1.f + __expf(-a));
  float w = wv[0];
  for (int c = 0; c < 3; ++c) eV[e * 3 + c] = Vh1[e * 3 + c] * w * g;
}

// ---------------------------------------------------------------------------
// message passing
// ---------------------------------------------------------------------------

__global__ void compute_deg(const int* __restrict__ dst, float* __restrict__ deg, int E) {
  int e = blockIdx.x * blockDim.x + threadIdx.x;
  if (e < E) atomicAdd(&deg[dst[e]], 1.f);
}

__global__ void msg_gather(const float* __restrict__ s, const float* __restrict__ es,
                           const int* __restrict__ src, const int* __restrict__ dst,
                           float* __restrict__ X, int E) {
  int i = blockIdx.x * blockDim.x + threadIdx.x;
  if (i >= E * 288) return;
  int e = i / 288, j = i - e * 288;
  float v;
  if (j < 128)      v = s[(size_t)src[e] * 128 + j];
  else if (j < 160) v = es[(size_t)e * 32 + (j - 128)];
  else              v = s[(size_t)dst[e] * 128 + (j - 160)];
  X[(size_t)e * 321 + j] = v;
}

// mV = [V[src](16), eV(1), V[dst](16)]; Vh = mV @ wh(33x33); X[:,288+h] = ||Vh||
__global__ void msg_vec(const float* __restrict__ V, const float* __restrict__ eV,
                        const int* __restrict__ src, const int* __restrict__ dst,
                        const float* __restrict__ wh,
                        float* __restrict__ Vh, float* __restrict__ X, int E) {
  int i = blockIdx.x * blockDim.x + threadIdx.x;
  if (i >= E * 33) return;
  int e = i / 33, h = i - e * 33;
  size_t se = (size_t)src[e] * 48, de = (size_t)dst[e] * 48;
  float a0 = 0, a1 = 0, a2 = 0;
  for (int v = 0; v < 33; ++v) {
    const float* p;
    if (v < 16)       p = V + se + v * 3;
    else if (v == 16) p = eV + (size_t)e * 3;
    else              p = V + de + (v - 17) * 3;
    float w = wh[v * 33 + h];
    a0 += p[0] * w; a1 += p[1] * w; a2 += p[2] * w;
  }
  size_t b = (size_t)e * 99 + h * 3;
  Vh[b] = a0; Vh[b + 1] = a1; Vh[b + 2] = a2;
  X[(size_t)e * 321 + 288 + h] = sqrtf(fmaxf(a0 * a0 + a1 * a1 + a2 * a2, EPSV));
}

__global__ void scatter_agg(const float* __restrict__ ms, const float* __restrict__ mV,
                            const int* __restrict__ dst,
                            float* __restrict__ aggs, float* __restrict__ aggv, int E) {
  int i = blockIdx.x * blockDim.x + threadIdx.x;
  if (i >= E * 128) return;
  int e = i / 128, j = i - e * 128;
  int d = dst[e];
  atomicAdd(&aggs[(size_t)d * 128 + j], ms[(size_t)e * 128 + j]);
  if (j < 48) atomicAdd(&aggv[(size_t)d * 48 + j], mV[(size_t)e * 48 + j]);
}

// s = LN(s + adds/deg), V = RMSnorm(V + addv/deg); null ptrs skip the residual
__global__ void residual_ln(float* __restrict__ s, float* __restrict__ V,
                            const float* __restrict__ adds, const float* __restrict__ addv,
                            const float* __restrict__ deg,
                            const float* __restrict__ lnb, const float* __restrict__ lng, int M) {
  int n = blockIdx.x * blockDim.x + threadIdx.x;
  if (n >= M) return;
  float sc = deg ? 1.f / fmaxf(deg[n], 1.f) : 1.f;
  size_t bs = (size_t)n * 128;
  float mu = 0;
  for (int j = 0; j < 128; ++j) {
    float x = s[bs + j] + (adds ? adds[bs + j] * sc : 0.f);
    s[bs + j] = x; mu += x;
  }
  mu *= (1.f / 128.f);
  float var = 0;
  for (int j = 0; j < 128; ++j) { float d = s[bs + j] - mu; var += d * d; }
  float rstd = rsqrtf(var * (1.f / 128.f) + 1e-5f);
  for (int j = 0; j < 128; ++j) s[bs + j] = (s[bs + j] - mu) * rstd * lng[j] + lnb[j];
  size_t bv = (size_t)n * 48;
  float ssum = 0;
  for (int v = 0; v < 16; ++v) {
    float ss = 0;
    for (int c = 0; c < 3; ++c) {
      float x = V[bv + v * 3 + c] + (addv ? addv[bv + v * 3 + c] * sc : 0.f);
      V[bv + v * 3 + c] = x; ss += x * x;
    }
    ssum += fmaxf(ss, EPSV);
  }
  float inv = rsqrtf(ssum * (1.f / 16.f));
  for (int j = 0; j < 48; ++j) V[bv + j] *= inv;
}

// ---------------------------------------------------------------------------
// readout / ensemble tail
// ---------------------------------------------------------------------------

__global__ void readout_acc(const float* __restrict__ out_n, const int* __restrict__ batch,
                            float* __restrict__ g, float* __restrict__ cnt, int M) {
  int i = blockIdx.x * blockDim.x + threadIdx.x;
  if (i >= M * 128) return;
  int n = i / 128, j = i - n * 128;
  int b = batch[n];
  atomicAdd(&g[(size_t)b * 128 + j], out_n[(size_t)n * 128 + j]);
  if (j == 0) atomicAdd(&cnt[b], 1.f);
}

__global__ void readout_div(const float* __restrict__ g, const float* __restrict__ cnt,
                            float* __restrict__ out, int NG) {
  int i = blockIdx.x * blockDim.x + threadIdx.x;
  if (i >= NG * 128) return;
  int b = i / 128;
  out[i] = g[i] / fmaxf(cnt[b], 1.f);
}

__global__ void fc2_k(const float* __restrict__ H, const float* __restrict__ w,
                      const float* __restrict__ b, float* __restrict__ out) {
  int i = blockIdx.x * blockDim.x + threadIdx.x;
  if (i >= 64) return;
  int g = i >> 1, o = i & 1;
  float a = b[o];
  for (int k = 0; k < 384; ++k) a += H[(size_t)g * 384 + k] * w[k * 2 + o];
  out[i] = a;
}

// ---------------------------------------------------------------------------
// host side
// ---------------------------------------------------------------------------

typedef struct { const float *wh, *wsb, *wsw, *wsvb, *wsvw, *wv; } GVPW;
typedef struct { const float *b, *g; } LNW;
typedef struct { GVPW ff0, ff1; LNW ln0, ln1; GVPW m0, m1, m2; } LayerW;
typedef struct {
  const float* W_s; GVPW We; LNW We_ln; GVPW Wout; LNW Wout_ln;
  GVPW Wv; LNW Wv_ln; const float *dense_b, *dense_w; LayerW L[3];
} EncW;

static const float* FPi(void* const* d, int& i) { return (const float*)d[i++]; }
static void rd_gvp(void* const* d, int& i, GVPW& g, bool full) {
  g.wh = FPi(d, i); g.wsb = FPi(d, i); g.wsw = FPi(d, i);
  if (full) { g.wsvb = FPi(d, i); g.wsvw = FPi(d, i); g.wv = FPi(d, i); }
  else { g.wsvb = g.wsvw = g.wv = nullptr; }
}
static void rd_ln(void* const* d, int& i, LNW& l) { l.b = FPi(d, i); l.g = FPi(d, i); }
static void rd_enc(void* const* d, int& i, EncW& e) {
  // sorted pytree keys: W_s, We, We_ln, Wout, Wout_ln, Wv, Wv_ln, dense, layers
  e.W_s = FPi(d, i);
  rd_gvp(d, i, e.We, true);  rd_ln(d, i, e.We_ln);
  rd_gvp(d, i, e.Wout, false); rd_ln(d, i, e.Wout_ln);
  rd_gvp(d, i, e.Wv, true);  rd_ln(d, i, e.Wv_ln);
  e.dense_b = FPi(d, i); e.dense_w = FPi(d, i);
  for (int l = 0; l < 3; ++l) {   // per-layer keys: ff, ln0, ln1, msg
    rd_gvp(d, i, e.L[l].ff0, true); rd_gvp(d, i, e.L[l].ff1, true);
    rd_ln(d, i, e.L[l].ln0); rd_ln(d, i, e.L[l].ln1);
    rd_gvp(d, i, e.L[l].m0, true); rd_gvp(d, i, e.L[l].m1, true); rd_gvp(d, i, e.L[l].m2, true);
  }
}

static void run_gemm(hipStream_t stream, _Float16* pk,
                     const float* W, const float* bias,
                     const float* A, int lda, float* C, int ldc,
                     int M, int K, int N, int act) {
  int Kp = CEILDIV(K, 32) * 32;
  pack_w<<<CEILDIV(N * Kp, 256), 256, 0, stream>>>(W, pk, K, N, Kp);
  if ((N & 31) == 0) {
    int tiles = CEILDIV(M, 16) * (N / 32);
    wmma_gemm_t<2><<<CEILDIV(tiles, 4), 128, 0, stream>>>(A, lda, pk, Kp, bias, C, ldc,
                                                          M, K, N, act);
  } else {
    int tiles = CEILDIV(M, 16) * (N / 16);
    wmma_gemm_t<1><<<CEILDIV(tiles, 4), 128, 0, stream>>>(A, lda, pk, Kp, bias, C, ldc,
                                                          M, K, N, act);
  }
}

struct WSPtrs {
  _Float16* pk;
  float *s, *V, *es, *eV, *deg;
  float *bufA, *bufB, *bufC, *bufD, *bufE, *bufF, *bufG;
  float *aggs, *aggv, *nVh, *nX, *nX2, *nV2, *gaten;
  float *gacc, *cnt, *gbuf, *H, *H2;
};

#define KL(kern, n, ...) kern<<<CEILDIV((n), 256), 256, 0, stream>>>(__VA_ARGS__)

static void run_encoder(hipStream_t stream, const EncW& E, WSPtrs& w,
                        const float* node_s, const float* node_v,
                        const float* edge_s, const float* edge_v,
                        const int* ei, const int* seq, const int* batch,
                        float* Hcol) {
  const int* src = ei;
  const int* dst = ei + NEDGES;
  const int Nn = NNODES, Ne = NEDGES;

  KL(fill_f, Nn, w.deg, 0.f, Nn);
  KL(compute_deg, Ne, dst, w.deg, Ne);

  // node embedding (GVP Wv)
  KL(node_embed_pre, Nn, node_s, node_v, seq, E.W_s, E.Wv_ln.b, E.Wv_ln.g, E.Wv.wh,
     w.nX, w.nVh, Nn);
  run_gemm(stream, w.pk, E.Wv.wsw, E.Wv.wsb, w.nX, 42, w.s, 128, Nn, 42, 128, 0);
  run_gemm(stream, w.pk, E.Wv.wsvw, E.Wv.wsvb, w.s, 128, w.gaten, 16, Nn, 128, 16, 2);
  KL(vec_out, Nn * 16, w.nVh, 16, w.gaten, E.Wv.wv, 16, w.V, Nn);

  // edge embedding (GVP We)
  KL(edge_embed_pre, Ne, edge_s, edge_v, E.We_ln.b, E.We_ln.g, E.We.wh, w.bufF, w.bufC, Ne);
  run_gemm(stream, w.pk, E.We.wsw, E.We.wsb, w.bufF, 33, w.es, 32, Ne, 33, 32, 0);
  KL(edge_embed_post, Ne, w.es, E.We.wsvw, E.We.wsvb, w.bufC, E.We.wv, w.eV, Ne);

  for (int l = 0; l < 3; ++l) {
    const LayerW& L = E.L[l];
    // ---- message GVP 0 ----
    KL(msg_gather, Ne * 288, w.s, w.es, src, dst, w.bufA, Ne);
    KL(msg_vec, Ne * 33, w.V, w.eV, src, dst, L.m0.wh, w.bufC, w.bufA, Ne);
    run_gemm(stream, w.pk, L.m0.wsw, L.m0.wsb, w.bufA, 321, w.bufF, 144, Ne, 321, 128, 0);
    run_gemm(stream, w.pk, L.m0.wsvw, L.m0.wsvb, w.bufF, 144, w.bufD, 16, Ne, 128, 16, 2);
    KL(vec_out, Ne * 16, w.bufC, 33, w.bufD, L.m0.wv, 16, w.bufE, Ne);
    KL(relu_cols, Ne * 128, w.bufF, 144, 128, Ne);
    // ---- message GVP 1 ----
    KL(vec_in, Ne * 16, w.bufE, 16, L.m1.wh, 16, w.bufC, w.bufF, 144, 128, Ne);
    run_gemm(stream, w.pk, L.m1.wsw, L.m1.wsb, w.bufF, 144, w.bufA, 144, Ne, 144, 128, 0);
    run_gemm(stream, w.pk, L.m1.wsvw, L.m1.wsvb, w.bufA, 144, w.bufD, 16, Ne, 128, 16, 2);
    KL(vec_out, Ne * 16, w.bufC, 16, w.bufD, L.m1.wv, 16, w.bufG, Ne);
    KL(relu_cols, Ne * 128, w.bufA, 144, 128, Ne);
    // ---- message GVP 2 (no scalar act) ----
    KL(vec_in, Ne * 16, w.bufG, 16, L.m2.wh, 16, w.bufC, w.bufA, 144, 128, Ne);
    run_gemm(stream, w.pk, L.m2.wsw, L.m2.wsb, w.bufA, 144, w.bufB, 128, Ne, 144, 128, 0);
    run_gemm(stream, w.pk, L.m2.wsvw, L.m2.wsvb, w.bufB, 128, w.bufD, 16, Ne, 128, 16, 2);
    KL(vec_out, Ne * 16, w.bufC, 16, w.bufD, L.m2.wv, 16, w.bufE, Ne);
    // ---- aggregate + residual LN ----
    KL(fill_f, Nn * 128, w.aggs, 0.f, Nn * 128);
    KL(fill_f, Nn * 48, w.aggv, 0.f, Nn * 48);
    KL(scatter_agg, Ne * 128, w.bufB, w.bufE, dst, w.aggs, w.aggv, Ne);
    KL(residual_ln, Nn, w.s, w.V, w.aggs, w.aggv, w.deg, L.ln0.b, L.ln0.g, Nn);
    // ---- feed-forward GVP 0 ----
    KL(copy_cols, Nn * 128, w.s, 128, w.nX, 160, 128, Nn);
    KL(vec_in, Nn * 32, w.V, 16, L.ff0.wh, 32, w.nVh, w.nX, 160, 128, Nn);
    run_gemm(stream, w.pk, L.ff0.wsw, L.ff0.wsb, w.nX, 160, w.nX2, 544, Nn, 160, 512, 0);
    run_gemm(stream, w.pk, L.ff0.wsvw, L.ff0.wsvb, w.nX2, 544, w.gaten, 32, Nn, 512, 32, 2);
    KL(vec_out, Nn * 32, w.nVh, 32, w.gaten, L.ff0.wv, 32, w.nV2, Nn);
    KL(relu_cols, Nn * 512, w.nX2, 544, 512, Nn);
    // ---- feed-forward GVP 1 ----
    KL(vec_in, Nn * 32, w.nV2, 32, L.ff1.wh, 32, w.nVh, w.nX2, 544, 512, Nn);
    run_gemm(stream, w.pk, L.ff1.wsw, L.ff1.wsb, w.nX2, 544, w.aggs, 128, Nn, 544, 128, 0);
    run_gemm(stream, w.pk, L.ff1.wsvw, L.ff1.wsvb, w.aggs, 128, w.gaten, 16, Nn, 128, 16, 2);
    KL(vec_out, Nn * 16, w.nVh, 32, w.gaten, L.ff1.wv, 16, w.aggv, Nn);
    KL(residual_ln, Nn, w.s, w.V, w.aggs, w.aggv, (const float*)nullptr, L.ln1.b, L.ln1.g, Nn);
  }

  // output GVP (vo=0) + readout + dense
  KL(residual_ln, Nn, w.s, w.V, (const float*)nullptr, (const float*)nullptr,
     (const float*)nullptr, E.Wout_ln.b, E.Wout_ln.g, Nn);
  KL(copy_cols, Nn * 128, w.s, 128, w.nX, 144, 128, Nn);
  KL(vec_in, Nn * 16, w.V, 16, E.Wout.wh, 16, w.nVh, w.nX, 144, 128, Nn);
  run_gemm(stream, w.pk, E.Wout.wsw, E.Wout.wsb, w.nX, 144, w.aggs, 128, Nn, 144, 128, 1);
  KL(fill_f, 32 * 128, w.gacc, 0.f, 32 * 128);
  KL(fill_f, 32, w.cnt, 0.f, 32);
  KL(readout_acc, Nn * 128, w.aggs, batch, w.gacc, w.cnt, Nn);
  KL(readout_div, 32 * 128, w.gacc, w.cnt, w.gbuf, 32);
  run_gemm(stream, w.pk, E.dense_w, E.dense_b, w.gbuf, 128, Hcol, 384, 32, 128, 128, 1);
}

extern "C" void kernel_launch(void* const* d_in, const int* in_sizes, int n_in,
                              void* d_out, int out_size, void* d_ws, size_t ws_size,
                              hipStream_t stream) {
  (void)in_sizes; (void)n_in; (void)out_size; (void)ws_size;

  // ---- params (pytree flattened in sorted-key order after 21 graph arrays) ----
  int ci = 21;
  const float* fc1b = FPi(d_in, ci); const float* fc1w = FPi(d_in, ci);
  const float* fc2b = FPi(d_in, ci); const float* fc2w = FPi(d_in, ci);
  EncW enc[3];
  for (int e = 0; e < 3; ++e) rd_enc(d_in, ci, enc[e]);

  // ---- workspace bump allocator ----
  float* base = (float*)d_ws;
  size_t off = 0;
  auto alloc = [&](size_t n) { float* p = base + off; off += n; return p; };
  WSPtrs w;
  w.pk   = (_Float16*)alloc(100000);               // 200k f16 (max pack = 384*384)
  w.s    = alloc((size_t)NNODES * 128);
  w.V    = alloc((size_t)NNODES * 48);
  w.es   = alloc((size_t)NEDGES * 32);
  w.eV   = alloc((size_t)NEDGES * 3);
  w.deg  = alloc(NNODES);
  w.bufA = alloc((size_t)NEDGES * 352);            // edge concat input (ld up to 321)
  w.bufB = alloc((size_t)NEDGES * 128);            // final ms
  w.bufC = alloc((size_t)NEDGES * 112);            // edge Vh (h up to 33)
  w.bufD = alloc((size_t)NEDGES * 32);             // edge gate
  w.bufE = alloc((size_t)NEDGES * 48);             // edge mV
  w.bufF = alloc((size_t)NEDGES * 144);            // edge X2
  w.bufG = alloc((size_t)NEDGES * 48);             // edge mV (ping-pong)
  w.aggs = alloc((size_t)NNODES * 128);
  w.aggv = alloc((size_t)NNODES * 48);
  w.nVh  = alloc((size_t)NNODES * 96);
  w.nX   = alloc((size_t)NNODES * 160);
  w.nX2  = alloc((size_t)NNODES * 544);
  w.nV2  = alloc((size_t)NNODES * 96);
  w.gaten= alloc((size_t)NNODES * 32);
  w.gacc = alloc(32 * 128);
  w.cnt  = alloc(32);
  w.gbuf = alloc(32 * 128);
  w.H    = alloc(32 * 384);
  w.H2   = alloc(32 * 384);
  (void)alloc(64);                                 // tail slack for GEMM over-reads

  // ---- three encoders ----
  for (int e = 0; e < 3; ++e) {
    const float* node_s = (const float*)d_in[7 * e + 0];
    const float* node_v = (const float*)d_in[7 * e + 1];
    const float* edge_s = (const float*)d_in[7 * e + 2];
    const float* edge_v = (const float*)d_in[7 * e + 3];
    const int*   ei     = (const int*)  d_in[7 * e + 4];
    const int*   seq    = (const int*)  d_in[7 * e + 5];
    const int*   batch  = (const int*)  d_in[7 * e + 6];
    run_encoder(stream, enc[e], w, node_s, node_v, edge_s, edge_v, ei, seq, batch,
                w.H + e * 128);
  }

  // ---- ensemble head: ELU(H @ FC1) @ FC2 ----
  run_gemm(stream, w.pk, fc1w, fc1b, w.H, 384, w.H2, 384, 32, 384, 384, 3);
  fc2_k<<<1, 64, 0, stream>>>(w.H2, fc2w, fc2b, (float*)d_out);
}
